// Model_DSTM_54906861912324
// MI455X (gfx1250) — matrix operand order repaired and checked by reading, not compiled, not run on hardware
//
#include <hip/hip_runtime.h>
#include <hip/hip_bf16.h>
#include <stdint.h>

// ---------------------------------------------------------------------------
// Problem constants (from reference): E=1024, H=256, L=2, NC=7, B=32, N=256
// ---------------------------------------------------------------------------
#define Bb   32
#define Nn   256
#define Ee   1024
#define Hh   256
#define NCc  7
#define Gg   512
#define M_ROWS (Bb * Nn)   // 8192

typedef __attribute__((ext_vector_type(16))) __bf16 v16bf;
typedef __attribute__((ext_vector_type(8)))  __bf16 v8bf;
typedef __attribute__((ext_vector_type(8)))  float  v8f;

__device__ __forceinline__ float sigmoidf_(float x) { return 1.f / (1.f + __expf(-x)); }

// ---------------------------------------------------------------------------
// Generic bf16 WMMA GEMM:  C[M,N] = act(A[M,K] @ B + bias)
// B supplied pre-transposed as Bt[N,K] so the B fragment is two contiguous
// 8-element K-runs per lane.  One wave (32 threads) per 16x16 output tile.
// act: 0=none 1=relu 2=tanh.  C (f32) and Cb (bf16 copy) are each optional.
// ---------------------------------------------------------------------------
__global__ void gemm_bf16_wmma(const __bf16* __restrict__ A,
                               const __bf16* __restrict__ Bt,
                               const float*  __restrict__ bias,
                               float*        __restrict__ C,
                               __bf16*       __restrict__ Cb,
                               int M, int Ncols, int K, int act)
{
    const int lane = threadIdx.x & 31;
    const int m0   = blockIdx.y * 16;
    const int n0   = blockIdx.x * 16;
    const int r    = lane & 15;        // row (A) / col (B) index within tile
    const int half = lane >> 4;        // lane half selects K sub-range
    const int kha  = half * 8;         // A: VGPR0-3 hold K=kh..kh+7, VGPR4-7 hold K=16+kh..
    const int khb  = half * 8;         // B: same K order as A (kh..kh+7, 16+kh..)

    const __bf16* arow = A  + (size_t)(m0 + r) * K;
    const __bf16* brow = Bt + (size_t)(n0 + r) * K;

    v8f acc = {};
    for (int k0 = 0; k0 < K; k0 += 32) {
        v8bf alo = *reinterpret_cast<const v8bf*>(arow + k0 + kha);
        v8bf ahi = *reinterpret_cast<const v8bf*>(arow + k0 + 16 + kha);
        v16bf a  = __builtin_shufflevector(alo, ahi,
                       0,1,2,3,4,5,6,7,8,9,10,11,12,13,14,15);
        v8bf blo = *reinterpret_cast<const v8bf*>(brow + k0 + khb);
        v8bf bhi = *reinterpret_cast<const v8bf*>(brow + k0 + 16 + khb);
        v16bf b  = __builtin_shufflevector(blo, bhi,
                       0,1,2,3,4,5,6,7,8,9,10,11,12,13,14,15);
        acc = __builtin_amdgcn_wmma_f32_16x16x32_bf16(
                  false, a, false, b, (short)0, acc, false, false);
    }

    const int   col = n0 + r;
    const float bv  = bias ? bias[col] : 0.f;
#pragma unroll
    for (int v = 0; v < 8; ++v) {
        int   row = m0 + v + half * 8;
        float o   = acc[v] + bv;
        if      (act == 1) o = fmaxf(o, 0.f);
        else if (act == 2) o = tanhf(o);
        if (C)  C [(size_t)row * Ncols + col] = o;
        if (Cb) Cb[(size_t)row * Ncols + col] = (__bf16)o;
    }
}

// ---------------------------------------------------------------------------
// Incremental causal graph attention, layer 0 only (layer 1 output is dead).
// One 256-thread workgroup per batch; prev rows are write-once so prev.wk
// dots ("kdot") are cached.  V-projection is factored:
//   out = (sum attn*sm*prev) @ Wr0 + (sum attn*(1-sm)*prev) @ Wr1
// ---------------------------------------------------------------------------
__global__ void gat_kernel(const float* __restrict__ x,
                           const unsigned char* __restrict__ adj,
                           const int*   __restrict__ s_mask,
                           const float* __restrict__ wq,
                           const float* __restrict__ wk,
                           const float* __restrict__ gb,     // gat_b, use [0]
                           const float* __restrict__ Wr0,    // [H,H] layer 0
                           const float* __restrict__ Wr1,
                           float*       __restrict__ prev)   // out: g1 [B,N,H]
{
    __shared__ float s_wk[Hh], s_qdot[Nn], s_kdot[Nn];
    __shared__ float s_w0[Nn], s_w1[Nn], s_a0[Hh], s_a1[Hh], s_red[Hh];

    const int t = threadIdx.x;
    const int b = blockIdx.x;
    const float* xb = x    + (size_t)b * Nn * Hh;
    float*       pb = prev + (size_t)b * Nn * Hh;

    s_wk[t] = wk[t];
    {   // qdot[i] = x[b,i,:].wq + bias   (thread t handles utterance i=t)
        float q = gb[0];
        const float* xr = xb + (size_t)t * Hh;
        for (int k = 0; k < Hh; ++k) q += xr[k] * wq[k];
        s_qdot[t] = q;
    }
    pb[t] = xb[t];                       // prev row 0 = x[:,0]
    __syncthreads();

    s_red[t] = xb[t] * s_wk[t];          // kdot[0]
    __syncthreads();
    for (int st = 128; st > 0; st >>= 1) {
        if (t < st) s_red[t] += s_red[t + st];
        __syncthreads();
    }
    if (t == 0) s_kdot[0] = s_red[0];
    __syncthreads();

    for (int i = 1; i < Nn; ++i) {
        // ---- masked attention logits over n < i ----
        float al = -3.0e38f;
        if (t < i) {
            al = s_kdot[t] + s_qdot[i];
            if (!adj[((size_t)b * Nn + i) * Nn + t]) al = -3.0e38f;
        }
        s_red[t] = al;
        __syncthreads();
        for (int st = 128; st > 0; st >>= 1) {
            if (t < st) s_red[t] = fmaxf(s_red[t], s_red[t + st]);
            __syncthreads();
        }
        const float mx = s_red[0];
        __syncthreads();

        float e = (t < i) ? __expf(al - mx) : 0.f;
        s_red[t] = e;
        __syncthreads();
        for (int st = 128; st > 0; st >>= 1) {
            if (t < st) s_red[t] += s_red[t + st];
            __syncthreads();
        }
        const float inv = 1.f / s_red[0];
        __syncthreads();

        float at  = e * inv;
        float smf = (t < i) ? (float)s_mask[((size_t)b * Nn + i) * Nn + t] : 0.f;
        s_w0[t] = (t < i) ? at * smf         : 0.f;
        s_w1[t] = (t < i) ? at * (1.f - smf) : 0.f;
        __syncthreads();

        // ---- factored weighted sums over prev rows (thread t = dim d) ----
        float a0 = 0.f, a1 = 0.f;
        for (int n = 0; n < i; ++n) {
            float p = pb[(size_t)n * Hh + t];
            a0 += s_w0[n] * p;
            a1 += s_w1[n] * p;
        }
        s_a0[t] = a0;
        s_a1[t] = a1;
        __syncthreads();

        // ---- out = a0 @ Wr0 + a1 @ Wr1 (columns coalesced across threads) --
        float o = 0.f;
        for (int k = 0; k < Hh; ++k)
            o += s_a0[k] * Wr0[(size_t)k * Hh + t]
               + s_a1[k] * Wr1[(size_t)k * Hh + t];
        pb[(size_t)i * Hh + t] = o;

        s_red[t] = o * s_wk[t];           // kdot[i]
        __syncthreads();
        for (int st = 128; st > 0; st >>= 1) {
            if (t < st) s_red[t] += s_red[t + st];
            __syncthreads();
        }
        if (t == 0) s_kdot[i] = s_red[0];
        __syncthreads();
    }
}

// ---------------------------------------------------------------------------
// Syn-LSTM scan: one workgroup per batch; h,c live on-chip; Uh/Uhm stream
// from L2 (resident; 1.25 MB).  xz/gm/gc are precomputed by WMMA GEMMs.
// ---------------------------------------------------------------------------
__global__ void lstm_kernel(const float* __restrict__ xz,   // [B,N,4H]
                            const float* __restrict__ gm,   // [B,N,H]
                            const float* __restrict__ gc,   // [B,N,H]
                            const float* __restrict__ Uh,   // [H,4H]
                            const float* __restrict__ Uhm,  // [H,H]
                            float*       __restrict__ hs)   // [B,N,H]
{
    __shared__ float s_h[Hh];
    const int t = threadIdx.x;
    const int b = blockIdx.x;
    float c = 0.f;
    s_h[t] = 0.f;
    __syncthreads();

    for (int step = 0; step < Nn; ++step) {
        const size_t base = (size_t)b * Nn + step;
        const float* xzr  = xz + base * (4 * Hh);
        float zi = xzr[t], zf = xzr[Hh + t], zo = xzr[2 * Hh + t], zu = xzr[3 * Hh + t];
        float mp = gm[base * Hh + t];
        for (int k = 0; k < Hh; ++k) {
            const float  hk = s_h[k];
            const float* up = Uh + (size_t)k * (4 * Hh);
            __builtin_prefetch(up + 4 * Hh, 0, 1);
            zi += hk * up[t];
            zf += hk * up[Hh + t];
            zo += hk * up[2 * Hh + t];
            zu += hk * up[3 * Hh + t];
            mp += hk * Uhm[(size_t)k * Hh + t];
        }
        const float ig = sigmoidf_(zi);
        const float fg = sigmoidf_(zf);
        const float og = sigmoidf_(zo);
        const float ug = tanhf(zu);
        const float m  = sigmoidf_(mp);
        c = fg * c + ig * ug + m * gc[base * Hh + t];
        const float h = og * tanhf(c);
        hs[base * Hh + t] = h;
        __syncthreads();
        s_h[t] = h;
        __syncthreads();
    }
}

// ---------------------------------------------------------------------------
// Small helper kernels
// ---------------------------------------------------------------------------
__global__ void f32_to_bf16_kernel(const float* __restrict__ in,
                                   __bf16* __restrict__ out, int n)
{
    int i = blockIdx.x * blockDim.x + threadIdx.x;
    if (i < n) out[i] = (__bf16)in[i];
}

// W[K,N] (row-major) -> Wt[N,K] bf16
__global__ void conv_t_kernel(const float* __restrict__ W,
                              __bf16* __restrict__ Wt, int K, int Ncols)
{
    int i = blockIdx.x * blockDim.x + threadIdx.x;
    if (i < K * Ncols) {
        int n = i / K, k = i % K;
        Wt[i] = (__bf16)W[(size_t)k * Ncols + n];
    }
}

// graph_features = concat(x, g1) along feature dim, cast to bf16
__global__ void gcat_kernel(const float* __restrict__ x,
                            const float* __restrict__ g1,
                            __bf16* __restrict__ gcat)
{
    int i = blockIdx.x * blockDim.x + threadIdx.x;   // M_ROWS * G threads
    if (i >= M_ROWS * Gg) return;
    int row = i >> 9, j = i & 511;
    float v = (j < Hh) ? x[(size_t)row * Hh + j] : g1[(size_t)row * Hh + (j - Hh)];
    gcat[i] = (__bf16)v;
}

// logits = y2 @ Wo + bo   (N=7, tiny -> VALU)
__global__ void logits_kernel(const float* __restrict__ y2,
                              const float* __restrict__ Wo,
                              const float* __restrict__ bo,
                              float* __restrict__ out)
{
    int i = blockIdx.x * blockDim.x + threadIdx.x;   // M_ROWS * NC
    if (i >= M_ROWS * NCc) return;
    int row = i / NCc, cc = i % NCc;
    float o = bo[cc];
    const float* yr = y2 + (size_t)row * Hh;
    for (int k = 0; k < Hh; ++k) o += yr[k] * Wo[k * NCc + cc];
    out[i] = o;
}

// ---------------------------------------------------------------------------
// Orchestration
// ---------------------------------------------------------------------------
extern "C" void kernel_launch(void* const* d_in, const int* in_sizes, int n_in,
                              void* d_out, int out_size, void* d_ws, size_t ws_size,
                              hipStream_t stream)
{
    (void)in_sizes; (void)n_in; (void)out_size; (void)ws_size;

    const float*         features = (const float*)d_in[0];
    const unsigned char* adj      = (const unsigned char*)d_in[1];
    const int*           s_mask   = (const int*)d_in[2];
    const float* We   = (const float*)d_in[5];
    const float* be   = (const float*)d_in[6];
    const float* wq   = (const float*)d_in[7];   // [L,H] -> layer 0 = first H
    const float* wk   = (const float*)d_in[8];
    const float* gb   = (const float*)d_in[9];   // [L]
    const float* Wr0  = (const float*)d_in[10];  // [L,H,H] -> layer 0 slice
    const float* Wr1  = (const float*)d_in[11];
    const float* Wx   = (const float*)d_in[12];
    const float* Uh   = (const float*)d_in[13];
    const float* bx   = (const float*)d_in[14];
    const float* Wgm  = (const float*)d_in[15];
    const float* Uhm  = (const float*)d_in[16];
    const float* bm   = (const float*)d_in[17];
    const float* Wgc  = (const float*)d_in[18];
    const float* bgc  = (const float*)d_in[19];
    const float* W1   = (const float*)d_in[20];
    const float* b1   = (const float*)d_in[21];
    const float* W2   = (const float*)d_in[22];
    const float* b2   = (const float*)d_in[23];
    const float* Wo   = (const float*)d_in[24];
    const float* bo   = (const float*)d_in[25];
    float* logits = (float*)d_out;

    // --- workspace bump allocator (256B aligned) ---
    char* ws = (char*)d_ws;
    size_t off = 0;
    auto alloc = [&](size_t bytes) -> char* {
        char* p = ws + off;
        off = (off + bytes + 255) & ~(size_t)255;
        return p;
    };
    __bf16* featb = (__bf16*)alloc((size_t)M_ROWS * Ee * 2);   // 16 MB
    __bf16* Wet   = (__bf16*)alloc((size_t)Hh * Ee * 2);
    __bf16* Wxt   = (__bf16*)alloc((size_t)(4 * Hh) * Ee * 2);
    __bf16* Wgmt  = (__bf16*)alloc((size_t)Hh * Gg * 2);
    __bf16* Wgct  = (__bf16*)alloc((size_t)Hh * Gg * 2);
    __bf16* W1t   = (__bf16*)alloc((size_t)Hh * Hh * 2);
    __bf16* W2t   = (__bf16*)alloc((size_t)Hh * Hh * 2);
    float*  xbuf  = (float*)alloc((size_t)M_ROWS * Hh * 4);    // 8 MB
    float*  g1    = (float*)alloc((size_t)M_ROWS * Hh * 4);    // 8 MB
    float*  xz    = (float*)alloc((size_t)M_ROWS * 4 * Hh * 4);// 32 MB
    float*  gmB   = (float*)alloc((size_t)M_ROWS * Hh * 4);
    float*  gcB   = (float*)alloc((size_t)M_ROWS * Hh * 4);
    float*  hs    = (float*)alloc((size_t)M_ROWS * Hh * 4);
    // aliases: featb is dead after the xz GEMM; xz is dead after the LSTM
    __bf16* gcat = (__bf16*)featb;                      // 8 MB  <= 16 MB
    __bf16* hsb  = (__bf16*)xz;                         // 4 MB
    __bf16* y1b  = (__bf16*)((char*)xz + (4u << 20));   // 4 MB
    float*  y2   = (float*) ((char*)xz + (8u << 20));   // 8 MB

    const int TPB = 256;

    // 1) bf16 conversions of A-matrix and (transposed) weights
    f32_to_bf16_kernel<<<(M_ROWS * Ee + TPB - 1) / TPB, TPB, 0, stream>>>(features, featb, M_ROWS * Ee);
    conv_t_kernel<<<(Ee * Hh       + TPB - 1) / TPB, TPB, 0, stream>>>(We,  Wet,  Ee, Hh);
    conv_t_kernel<<<(Ee * 4 * Hh   + TPB - 1) / TPB, TPB, 0, stream>>>(Wx,  Wxt,  Ee, 4 * Hh);
    conv_t_kernel<<<(Gg * Hh       + TPB - 1) / TPB, TPB, 0, stream>>>(Wgm, Wgmt, Gg, Hh);
    conv_t_kernel<<<(Gg * Hh       + TPB - 1) / TPB, TPB, 0, stream>>>(Wgc, Wgct, Gg, Hh);
    conv_t_kernel<<<(Hh * Hh       + TPB - 1) / TPB, TPB, 0, stream>>>(W1,  W1t,  Hh, Hh);
    conv_t_kernel<<<(Hh * Hh       + TPB - 1) / TPB, TPB, 0, stream>>>(W2,  W2t,  Hh, Hh);

    // 2) x = relu(features @ We + be)   [8192,256] via WMMA
    gemm_bf16_wmma<<<dim3(Hh / 16, M_ROWS / 16), 32, 0, stream>>>(
        featb, Wet, be, xbuf, nullptr, M_ROWS, Hh, Ee, 1);

    // 3) xz = features @ Wx + bx        [8192,1024] via WMMA (17 GFLOP)
    gemm_bf16_wmma<<<dim3(4 * Hh / 16, M_ROWS / 16), 32, 0, stream>>>(
        featb, Wxt, bx, xz, nullptr, M_ROWS, 4 * Hh, Ee, 0);

    // 4) GAT layer 0 (layer 1 output is dead code in the reference)
    gat_kernel<<<Bb, 256, 0, stream>>>(xbuf, adj, s_mask, wq, wk, gb,
                                       Wr0, Wr1, g1);

    // 5) graph_features = concat(x, g1) in bf16 (aliases featb)
    gcat_kernel<<<(M_ROWS * Gg + TPB - 1) / TPB, TPB, 0, stream>>>(xbuf, g1, gcat);

    // 6) gm = g @ Wgm + bm ; gc = tanh(g @ Wgc + bgc)   via WMMA
    gemm_bf16_wmma<<<dim3(Hh / 16, M_ROWS / 16), 32, 0, stream>>>(
        gcat, Wgmt, bm, gmB, nullptr, M_ROWS, Hh, Gg, 0);
    gemm_bf16_wmma<<<dim3(Hh / 16, M_ROWS / 16), 32, 0, stream>>>(
        gcat, Wgct, bgc, gcB, nullptr, M_ROWS, Hh, Gg, 2);

    // 7) Syn-LSTM scan
    lstm_kernel<<<Bb, 256, 0, stream>>>(xz, gmB, gcB, Uh, Uhm, hs);

    // 8) MLP head: y1 = relu(hs@W1+b1) (bf16 only), y2 = relu(y1@W2+b2)
    f32_to_bf16_kernel<<<(M_ROWS * Hh + TPB - 1) / TPB, TPB, 0, stream>>>(hs, hsb, M_ROWS * Hh);
    gemm_bf16_wmma<<<dim3(Hh / 16, M_ROWS / 16), 32, 0, stream>>>(
        hsb, W1t, b1, nullptr, y1b, M_ROWS, Hh, Hh, 1);
    gemm_bf16_wmma<<<dim3(Hh / 16, M_ROWS / 16), 32, 0, stream>>>(
        y1b, W2t, b2, y2, nullptr, M_ROWS, Hh, Hh, 1);

    // 9) logits = y2 @ Wo + bo
    logits_kernel<<<(M_ROWS * NCc + TPB - 1) / TPB, TPB, 0, stream>>>(y2, Wo, bo, logits);
}
